// HFAM_37546604102406
// MI455X (gfx1250) — matrix operand order repaired and checked by reading, not compiled, not run on hardware
//
#include <hip/hip_runtime.h>
#include <hip/hip_bf16.h>
#include <math.h>

typedef __attribute__((ext_vector_type(16))) __bf16 v16bf;
typedef __attribute__((ext_vector_type(8)))  __bf16 v8bf;
typedef __attribute__((ext_vector_type(8)))  float  v8f;

#define N_NODES   50000
#define N_EDGES   600000
#define NUM_GRAPHS 256
#define NUM_TASK   8
#define IN_F       16
#define HID        128
#define OUT_F      256
#define HEAD_D     64

#define EPI_RELU   1
#define EPI_SKIP   2

__device__ __forceinline__ float gelu_exact(float x) {
    return 0.5f * x * (1.0f + erff(x * 0.70710678118654752440f));
}

// ---------------------------------------------------------------------------
// bf16 WMMA GEMM, no guards in the hot loop.
//   A : bf16 row-major, lda elements per row (lda % 32 == 0, zero-padded)
//   Wt: bf16 "transposed weights": Wt[col * KTOT + k] = W[k][col], zero-padded
//   C : fp32 row-major
// One wave computes one 16x16 tile. K loop fully unrolled (KTOT template).
// Fragment layouts used here (A/B k order is contiguous, same for both):
//   A  (16x32, 16-bit): lane half h, elems i: K = h*16 + i
//                       -> two contiguous 8-element chunks per step
//   B  (32x16, 16-bit): lane half h, elems i: K = h*16 + i -> 16 contiguous
//   C/D (16x16 f32):    row = r + h*8, col = lane&15
// ---------------------------------------------------------------------------
template<int KTOT>
__global__ __launch_bounds__(32)
void wmma_gemm_bf16(const __bf16* __restrict__ A, int lda, int sAz,
                    const __bf16* __restrict__ Wt, int sWz,
                    float* __restrict__ C, int ldc, int sCz,
                    const float* __restrict__ bias,
                    int flags,
                    const float* __restrict__ skipScalar,
                    const float* __restrict__ Xprev, int ldx)
{
    const int lane = threadIdx.x & 31;
    const int half = lane >> 4;
    const int r16  = lane & 15;
    const int m0   = blockIdx.x * 16;
    const int n0   = blockIdx.y * 16;
    const int z    = blockIdx.z;
    A  += (size_t)z * sAz;
    Wt += (size_t)z * sWz;
    C  += (size_t)z * sCz;

    const __bf16* Arow = A  + (size_t)(m0 + r16) * lda + (half << 4);
    const __bf16* Wcol = Wt + (size_t)(n0 + r16) * KTOT + (half << 4);
    const int col = n0 + r16;

    v8f acc = {};
#pragma unroll
    for (int k0 = 0; k0 < KTOT; k0 += 32) {
        v8bf alo = *(const v8bf*)(Arow + k0);
        v8bf ahi = *(const v8bf*)(Arow + k0 + 8);
        v16bf afrag = __builtin_shufflevector(alo, ahi,
                        0,1,2,3,4,5,6,7,8,9,10,11,12,13,14,15);
        v16bf bfrag = *(const v16bf*)(Wcol + k0);
        acc = __builtin_amdgcn_wmma_f32_16x16x32_bf16(false, afrag, false, bfrag,
                                                      (short)0, acc, false, false);
    }

    float g = 0.0f;
    if (flags & EPI_SKIP) g = 1.0f / (1.0f + expf(-skipScalar[0]));
#pragma unroll
    for (int r = 0; r < 8; ++r) {
        int row = m0 + r + (half << 3);
        float v = acc[r];
        if (bias) v += bias[col];
        if (flags & EPI_SKIP) v = g * v + (1.0f - g) * Xprev[(size_t)row * ldx + col];
        if (flags & EPI_RELU) v = fmaxf(v, 0.0f);
        C[(size_t)row * ldc + col] = v;
    }
}

// ---------------------------------------------------------------------------
// Prep kernels: fp32 -> bf16 with zero padding (and optional fused GELU),
// and weight transpose fp32 [Z][K][N] -> bf16 [Z][N][Kpad].
// ---------------------------------------------------------------------------
__global__ void rows_to_bf16(const float* __restrict__ src, int srcld, int srcK,
                             __bf16* __restrict__ dst, int dstld, int gelu, int total)
{
    int i = blockIdx.x * blockDim.x + threadIdx.x;
    if (i >= total) return;
    int k   = i % dstld;
    int row = i / dstld;
    float v = (k < srcK) ? src[(size_t)row * srcld + k] : 0.0f;
    if (gelu) v = gelu_exact(v);
    dst[i] = (__bf16)v;
}

__global__ void wt_to_bf16(const float* __restrict__ w, int K, int N,
                           __bf16* __restrict__ wt, int Kpad, int total)
{
    int i = blockIdx.x * blockDim.x + threadIdx.x;
    if (i >= total) return;
    int k = i % Kpad;
    int t = i / Kpad;
    int n = t % N;
    int z = t / N;
    wt[i] = (__bf16)((k < K) ? w[(size_t)z * K * N + (size_t)k * N + n] : 0.0f);
}

// ---------------------------------------------------------------------------
// Edge-phase kernels (wave-per-edge)
// ---------------------------------------------------------------------------
__device__ __forceinline__ void atomicMaxFloat(float* addr, float v) {
    if (v >= 0.0f) atomicMax((int*)addr, __float_as_int(v));
    else           atomicMin((unsigned int*)addr, __float_as_uint(v));
}

__global__ void edge_alpha_max(const int* __restrict__ src, const int* __restrict__ dst,
                               const float* __restrict__ q, const float* __restrict__ kr,
                               const float* __restrict__ p_rel,
                               float* __restrict__ alpha, float* __restrict__ smax, int E)
{
    int e    = blockIdx.x * (blockDim.x >> 5) + (threadIdx.x >> 5);
    int lane = threadIdx.x & 31;
    if (e >= E) return;
    int s = src[e], d = dst[e];
    const float* qd = q  + (size_t)d * HID;
    const float* ks = kr + (size_t)s * HID;
    float a0 = qd[lane]      * ks[lane]      + qd[lane + 32] * ks[lane + 32];
    float a1 = qd[lane + 64] * ks[lane + 64] + qd[lane + 96] * ks[lane + 96];
#pragma unroll
    for (int off = 16; off > 0; off >>= 1) {
        a0 += __shfl_xor(a0, off, 32);
        a1 += __shfl_xor(a1, off, 32);
    }
    if (lane == 0) {
        a0 *= p_rel[0] * 0.125f;   // 1/sqrt(64)
        a1 *= p_rel[1] * 0.125f;
        alpha[(size_t)e * 2]     = a0;
        alpha[(size_t)e * 2 + 1] = a1;
        atomicMaxFloat(&smax[(size_t)d * 2],     a0);
        atomicMaxFloat(&smax[(size_t)d * 2 + 1], a1);
    }
}

__global__ void edge_exp_sum(const int* __restrict__ dst, float* __restrict__ alpha,
                             const float* __restrict__ smax, float* __restrict__ ssum, int E)
{
    int t = blockIdx.x * blockDim.x + threadIdx.x;
    if (t >= E * 2) return;
    int e = t >> 1, h = t & 1;
    int d = dst[e];
    float ev = __expf(alpha[t] - smax[(size_t)d * 2 + h]);
    alpha[t] = ev;
    atomicAdd(&ssum[(size_t)d * 2 + h], ev);
}

__global__ void edge_aggregate(const int* __restrict__ src, const int* __restrict__ dst,
                               const float* __restrict__ alpha, const float* __restrict__ ssum,
                               const float* __restrict__ vr, float* __restrict__ out, int E)
{
    int e    = blockIdx.x * (blockDim.x >> 5) + (threadIdx.x >> 5);
    int lane = threadIdx.x & 31;
    if (e >= E) return;
    int s = src[e], d = dst[e];
    float a0 = alpha[(size_t)e * 2]     / ssum[(size_t)d * 2];
    float a1 = alpha[(size_t)e * 2 + 1] / ssum[(size_t)d * 2 + 1];
    const float* vs = vr  + (size_t)s * HID;
    float*       od = out + (size_t)d * HID;
    atomicAdd(&od[lane],      a0 * vs[lane]);
    atomicAdd(&od[lane + 32], a0 * vs[lane + 32]);
    atomicAdd(&od[lane + 64], a1 * vs[lane + 64]);
    atomicAdd(&od[lane + 96], a1 * vs[lane + 96]);
}

// ---------------------------------------------------------------------------
__global__ void pool_mask(const float* __restrict__ x, const int* __restrict__ batch,
                          const float* __restrict__ task, const float* __restrict__ h,
                          float* __restrict__ pooled, int N)
{
    int n    = blockIdx.x * (blockDim.x >> 5) + (threadIdx.x >> 5);
    int lane = threadIdx.x & 31;
    if (n >= N) return;
    int b = batch[n];
    float cls = x[(size_t)n * IN_F + 7];
    bool hit = false;
#pragma unroll
    for (int t = 0; t < NUM_TASK; ++t) hit = hit || (cls == task[b * NUM_TASK + t]);
    if (!hit) return;
    const float* hn = h + (size_t)n * HID;
    float* pb = pooled + (size_t)b * HID;
#pragma unroll
    for (int j = 0; j < HID; j += 32) atomicAdd(&pb[j + lane], hn[j + lane]);
}

__global__ void fill_f32(float* __restrict__ p, float v, int n) {
    int i = blockIdx.x * blockDim.x + threadIdx.x;
    if (i < n) p[i] = v;
}

// ---------------------------------------------------------------------------
extern "C" void kernel_launch(void* const* d_in, const int* in_sizes, int n_in,
                              void* d_out, int out_size, void* d_ws, size_t ws_size,
                              hipStream_t stream) {
    (void)in_sizes; (void)n_in; (void)out_size; (void)ws_size;

    const float* x     = (const float*)d_in[0];
    const int*   ei0   = (const int*)d_in[1];   // [2, E]
    const int*   ei1   = (const int*)d_in[2];
    const int*   batch = (const int*)d_in[3];
    const float* task  = (const float*)d_in[4];

    struct Conv {
        const float *kw, *kb, *qw, *qb, *vw, *vb, *aw, *ab, *skip;
        const float *arel[2], *mrel[2], *prel[2];
    };
    auto getConv = [&](int base) {
        Conv c;
        c.kw = (const float*)d_in[base + 0];  c.kb = (const float*)d_in[base + 1];
        c.qw = (const float*)d_in[base + 2];  c.qb = (const float*)d_in[base + 3];
        c.vw = (const float*)d_in[base + 4];  c.vb = (const float*)d_in[base + 5];
        c.aw = (const float*)d_in[base + 6];  c.ab = (const float*)d_in[base + 7];
        c.skip = (const float*)d_in[base + 8];
        c.arel[0] = (const float*)d_in[base + 9];
        c.mrel[0] = (const float*)d_in[base + 10];
        c.prel[0] = (const float*)d_in[base + 11];
        c.arel[1] = (const float*)d_in[base + 12];
        c.mrel[1] = (const float*)d_in[base + 13];
        c.prel[1] = (const float*)d_in[base + 14];
        return c;
    };
    Conv conv[3] = { getConv(5), getConv(20), getConv(35) };
    const float* w1 = (const float*)d_in[50];
    const float* b1 = (const float*)d_in[51];
    const float* w2 = (const float*)d_in[52];
    const float* b2 = (const float*)d_in[53];

    // ---- workspace carve -------------------------------------------------
    char* wsb = (char*)d_ws;
    size_t off = 0;
    auto carve = [&](size_t bytes) -> char* {
        char* p = wsb + off;
        off = (off + bytes + 255) & ~(size_t)255;
        return p;
    };
    const size_t NF = (size_t)N_NODES * HID;
    // fp32 buffers
    float* hA    = (float*)carve(NF * 4);
    float* hB    = (float*)carve(NF * 4);
    float* kf    = (float*)carve(NF * 4);
    float* qf    = (float*)carve(NF * 4);
    float* vf    = (float*)carve(NF * 4);
    float* krf   = (float*)carve(NF * 4);
    float* vrf   = (float*)carve(NF * 4);
    float* aggf  = (float*)carve(NF * 4);
    float* alpha = (float*)carve((size_t)N_EDGES * 2 * 4);
    float* smax  = (float*)carve((size_t)N_NODES * 2 * 4);
    float* ssum  = (float*)carve((size_t)N_NODES * 2 * 4);
    float* pooled= (float*)carve((size_t)NUM_GRAPHS * HID * 4);
    float* hid   = (float*)carve((size_t)NUM_GRAPHS * HID * 4);
    // bf16 buffers
    __bf16* xb    = (__bf16*)carve(NF * 2);                  // layer input (padded)
    __bf16* kb    = (__bf16*)carve(NF * 2);
    __bf16* vb    = (__bf16*)carve(NF * 2);
    __bf16* aggb  = (__bf16*)carve(NF * 2);
    __bf16* pooledb = (__bf16*)carve((size_t)NUM_GRAPHS * HID * 2);
    __bf16* hidb    = (__bf16*)carve((size_t)NUM_GRAPHS * HID * 2);
    __bf16* kwt  = (__bf16*)carve((size_t)HID * HID * 2);
    __bf16* qwt  = (__bf16*)carve((size_t)HID * HID * 2);
    __bf16* vwt  = (__bf16*)carve((size_t)HID * HID * 2);
    __bf16* awt  = (__bf16*)carve((size_t)HID * HID * 2);
    __bf16* arelT= (__bf16*)carve((size_t)2 * HEAD_D * HEAD_D * 2);
    __bf16* mrelT= (__bf16*)carve((size_t)2 * HEAD_D * HEAD_D * 2);
    __bf16* w1t  = (__bf16*)carve((size_t)HID * HID * 2);
    __bf16* w2t  = (__bf16*)carve((size_t)OUT_F * HID * 2);

    const int MT = N_NODES / 16;           // 3125 row tiles (exact)
    dim3 w32(32, 1, 1);
    dim3 b256(256, 1, 1);
    const int edgeWaveBlocks = (N_EDGES + 7) / 8;
    const int* eix[2] = { ei0, ei1 };
    auto eb = [](int n) { return (n + 255) / 256; };

    const float* Xin = x;
    int ldx_in = IN_F, Kin = IN_F;
    float* houts[3] = { hA, hB, hA };

    for (int L = 0; L < 3; ++L) {
        const Conv& c = conv[L];
        const int Kpad = (Kin + 31) & ~31;   // 32 (L0) or 128

        // layer input -> bf16 padded
        int totX = N_NODES * Kpad;
        rows_to_bf16<<<eb(totX), b256, 0, stream>>>(Xin, ldx_in, Kin, xb, Kpad, 0, totX);
        // projection weights -> transposed bf16
        int totW = HID * Kpad;
        wt_to_bf16<<<eb(totW), b256, 0, stream>>>(c.kw, Kin, HID, kwt, Kpad, totW);
        wt_to_bf16<<<eb(totW), b256, 0, stream>>>(c.qw, Kin, HID, qwt, Kpad, totW);
        wt_to_bf16<<<eb(totW), b256, 0, stream>>>(c.vw, Kin, HID, vwt, Kpad, totW);

        // k/q/v projections
        if (Kpad == 32) {
            wmma_gemm_bf16<32><<<dim3(MT, HID/16, 1), w32, 0, stream>>>(
                xb, 32, 0, kwt, 0, kf, HID, 0, c.kb, 0, nullptr, nullptr, 0);
            wmma_gemm_bf16<32><<<dim3(MT, HID/16, 1), w32, 0, stream>>>(
                xb, 32, 0, qwt, 0, qf, HID, 0, c.qb, 0, nullptr, nullptr, 0);
            wmma_gemm_bf16<32><<<dim3(MT, HID/16, 1), w32, 0, stream>>>(
                xb, 32, 0, vwt, 0, vf, HID, 0, c.vb, 0, nullptr, nullptr, 0);
        } else {
            wmma_gemm_bf16<128><<<dim3(MT, HID/16, 1), w32, 0, stream>>>(
                xb, 128, 0, kwt, 0, kf, HID, 0, c.kb, 0, nullptr, nullptr, 0);
            wmma_gemm_bf16<128><<<dim3(MT, HID/16, 1), w32, 0, stream>>>(
                xb, 128, 0, qwt, 0, qf, HID, 0, c.qb, 0, nullptr, nullptr, 0);
            wmma_gemm_bf16<128><<<dim3(MT, HID/16, 1), w32, 0, stream>>>(
                xb, 128, 0, vwt, 0, vf, HID, 0, c.vb, 0, nullptr, nullptr, 0);
        }

        // k,v -> bf16 for the relation transforms
        int totN = N_NODES * HID;
        rows_to_bf16<<<eb(totN), b256, 0, stream>>>(kf, HID, HID, kb, HID, 0, totN);
        rows_to_bf16<<<eb(totN), b256, 0, stream>>>(vf, HID, HID, vb, HID, 0, totN);

        fill_f32<<<eb(totN), b256, 0, stream>>>(aggf, 0.0f, totN);

        for (int r = 0; r < 2; ++r) {
            // relation weights [2,64,64] -> transposed bf16 per head
            int totR = 2 * HEAD_D * HEAD_D;
            wt_to_bf16<<<eb(totR), b256, 0, stream>>>(c.arel[r], HEAD_D, HEAD_D, arelT, HEAD_D, totR);
            wt_to_bf16<<<eb(totR), b256, 0, stream>>>(c.mrel[r], HEAD_D, HEAD_D, mrelT, HEAD_D, totR);

            // per-head: k_r = k[:,h] @ a_rel[h], v_r = v[:,h] @ m_rel[h]
            wmma_gemm_bf16<64><<<dim3(MT, HEAD_D/16, 2), w32, 0, stream>>>(
                kb, HID, HEAD_D, arelT, HEAD_D*HEAD_D, krf, HID, HEAD_D,
                nullptr, 0, nullptr, nullptr, 0);
            wmma_gemm_bf16<64><<<dim3(MT, HEAD_D/16, 2), w32, 0, stream>>>(
                vb, HID, HEAD_D, mrelT, HEAD_D*HEAD_D, vrf, HID, HEAD_D,
                nullptr, 0, nullptr, nullptr, 0);

            int n2 = N_NODES * 2;
            fill_f32<<<eb(n2), b256, 0, stream>>>(smax, -INFINITY, n2);
            fill_f32<<<eb(n2), b256, 0, stream>>>(ssum, 0.0f, n2);

            const int* srcp = eix[r];
            const int* dstp = eix[r] + N_EDGES;
            edge_alpha_max<<<edgeWaveBlocks, b256, 0, stream>>>(
                srcp, dstp, qf, krf, c.prel[r], alpha, smax, N_EDGES);
            edge_exp_sum<<<eb(2 * N_EDGES), b256, 0, stream>>>(
                dstp, alpha, smax, ssum, N_EDGES);
            edge_aggregate<<<edgeWaveBlocks, b256, 0, stream>>>(
                srcp, dstp, alpha, ssum, vrf, aggf, N_EDGES);
        }

        // gelu(agg) -> bf16, then @ a_w with fused bias/skip/relu epilogue
        rows_to_bf16<<<eb(totN), b256, 0, stream>>>(aggf, HID, HID, aggb, HID, 1, totN);
        int totA = HID * HID;
        wt_to_bf16<<<eb(totA), b256, 0, stream>>>(c.aw, HID, HID, awt, HID, totA);
        int flags = EPI_RELU | (L > 0 ? EPI_SKIP : 0);
        wmma_gemm_bf16<128><<<dim3(MT, HID/16, 1), w32, 0, stream>>>(
            aggb, HID, 0, awt, 0, houts[L], HID, 0, c.ab,
            flags, c.skip, Xin, ldx_in);

        Xin = houts[L]; ldx_in = HID; Kin = HID;
    }

    // masked pooling
    int np = NUM_GRAPHS * HID;
    fill_f32<<<eb(np), b256, 0, stream>>>(pooled, 0.0f, np);
    pool_mask<<<(N_NODES + 7) / 8, b256, 0, stream>>>(x, batch, task, Xin, pooled, N_NODES);

    // MLP: relu(pooled @ w1 + b1) @ w2 + b2
    rows_to_bf16<<<eb(np), b256, 0, stream>>>(pooled, HID, HID, pooledb, HID, 0, np);
    int totW1 = HID * HID;
    wt_to_bf16<<<eb(totW1), b256, 0, stream>>>(w1, HID, HID, w1t, HID, totW1);
    wmma_gemm_bf16<128><<<dim3(NUM_GRAPHS/16, HID/16, 1), w32, 0, stream>>>(
        pooledb, HID, 0, w1t, 0, hid, HID, 0, b1, EPI_RELU, nullptr, nullptr, 0);

    rows_to_bf16<<<eb(np), b256, 0, stream>>>(hid, HID, HID, hidb, HID, 0, np);
    int totW2 = OUT_F * HID;
    wt_to_bf16<<<eb(totW2), b256, 0, stream>>>(w2, HID, OUT_F, w2t, HID, totW2);
    wmma_gemm_bf16<128><<<dim3(NUM_GRAPHS/16, OUT_F/16, 1), w32, 0, stream>>>(
        hidb, HID, 0, w2t, 0, (float*)d_out, OUT_F, 0, b2, 0, nullptr, nullptr, 0);
}
